// EdgeEmbAttentionAggregator_46033459479179
// MI455X (gfx1250) — matrix operand order repaired and checked by reading, not compiled, not run on hardware
//
#include <hip/hip_runtime.h>
#include <hip/hip_bf16.h>

typedef __attribute__((ext_vector_type(16))) __bf16 v16bf;
typedef __attribute__((ext_vector_type(4)))  __bf16 v4bf;
typedef __attribute__((ext_vector_type(2)))  __bf16 v2bf;
typedef __attribute__((ext_vector_type(8)))  float  v8f;

__device__ __forceinline__ float elu_f(float v) {
    return v > 0.f ? v : (__expf(v) - 1.f);
}

// -------------------------------------------------------------------------
// Kernel 1: x = input @ W  (bf16 WMMA, f32 accum)
//   - writes elu(x) to out[:, 0:128]
//   - writes s_x = x @ a_x to workspace (per node)
// 128 threads = 4 waves; each wave owns one 16-row m-tile; 64 nodes / block.
// -------------------------------------------------------------------------
__global__ __launch_bounds__(128) void gat_x_kernel(
    const float* __restrict__ input, const float* __restrict__ W,
    const float* __restrict__ a, float* __restrict__ out,
    float* __restrict__ sx, int N)
{
    __shared__ __bf16 wT[128 * 128];   // W transposed: wT[n*128 + k] = W[k][n]
    __shared__ float  ax[128];
    __shared__ float  sxl[64];

    const int tid  = threadIdx.x;
    const int wave = tid >> 5;
    const int lane = tid & 31;
    const int half = lane >> 4;
    const int nl   = lane & 15;

    // cooperative transpose W (f32 -> bf16); coalesced global reads
    for (int i = 0; i < 128; ++i)
        wT[tid * 128 + i] = (__bf16)W[i * 128 + tid];
    if (tid < 128) ax[tid]  = a[tid];
    if (tid < 64)  sxl[tid] = 0.f;
    __syncthreads();

    const int mt_g  = blockIdx.x * 4 + wave;    // global 16-node tile
    const int rbase = mt_g * 16;
    // uniform (SGPR) tail test -> single scalar branch, clean store path
    const bool fullTile = (__builtin_amdgcn_readfirstlane(rbase) + 16) <= N;

    // A fragments: lane holds row m = lane&15; K contiguous, lane half h holds k = 16h..16h+15 of each 32-k step
    int arow = rbase + nl; if (arow >= N) arow = N - 1;   // clamp (loads only)
    const float* ap = input + (size_t)arow * 128;
    v16bf afrag[4];
#pragma unroll
    for (int kk = 0; kk < 4; ++kk) {
        const int kb = kk * 32 + half * 16;
        const float4 q0 = *(const float4*)(ap + kb);
        const float4 q1 = *(const float4*)(ap + kb + 4);
        const float4 q2 = *(const float4*)(ap + kb + 8);
        const float4 q3 = *(const float4*)(ap + kb + 12);
        afrag[kk][0]  = (__bf16)q0.x; afrag[kk][1]  = (__bf16)q0.y;
        afrag[kk][2]  = (__bf16)q0.z; afrag[kk][3]  = (__bf16)q0.w;
        afrag[kk][4]  = (__bf16)q1.x; afrag[kk][5]  = (__bf16)q1.y;
        afrag[kk][6]  = (__bf16)q1.z; afrag[kk][7]  = (__bf16)q1.w;
        afrag[kk][8]  = (__bf16)q2.x; afrag[kk][9]  = (__bf16)q2.y;
        afrag[kk][10] = (__bf16)q2.z; afrag[kk][11] = (__bf16)q2.w;
        afrag[kk][12] = (__bf16)q3.x; afrag[kk][13] = (__bf16)q3.y;
        afrag[kk][14] = (__bf16)q3.z; afrag[kk][15] = (__bf16)q3.w;
    }

    float spart[8];
#pragma unroll
    for (int r = 0; r < 8; ++r) spart[r] = 0.f;

    for (int nt = 0; nt < 8; ++nt) {
        v8f acc = {};
        const __bf16* bp = &wT[(nt * 16 + nl) * 128 + half * 16];
#pragma unroll
        for (int kk = 0; kk < 4; ++kk) {
            v16bf bfrag = *(const v16bf*)(bp + kk * 32);
            acc = __builtin_amdgcn_wmma_f32_16x16x32_bf16(
                false, afrag[kk], false, bfrag, (short)0, acc, false, false);
        }
        const int   col  = nt * 16 + nl;
        const float acol = ax[col];
        if (fullTile) {
#pragma unroll
            for (int r = 0; r < 8; ++r) {
                const float v = acc[r];
                spart[r] += v * acol;
                out[(size_t)(rbase + r + 8 * half) * 320 + col] = elu_f(v);
            }
        } else {
#pragma unroll
            for (int r = 0; r < 8; ++r) {
                const int   row = rbase + r + 8 * half;
                const float v   = acc[r];
                spart[r] += v * acol;
                if (row < N) out[(size_t)row * 320 + col] = elu_f(v);
            }
        }
    }
#pragma unroll
    for (int r = 0; r < 8; ++r)
        atomicAdd(&sxl[wave * 16 + r + 8 * half], spart[r]);   // ds_add_f32
    __syncthreads();
    if (tid < 64) {
        const int row = blockIdx.x * 64 + tid;
        if (row < N) sx[row] = sxl[tid];
    }
}

// -------------------------------------------------------------------------
// Kernel 2: fused neighbor pipeline, 8 nodes (=256 neighbor rows) / block.
//   neighs = neigh_feat @ W2 via bf16 WMMA, kept in LDS (bf16, never to HBM)
//   scores -> leakyReLU -> softmax -> h_prime / e_agg -> elu -> out[:,128:320]
// 256 threads = 8 waves; each wave does 2 m-tiles x 8 n-tiles x 4 K-steps.
// Dynamic LDS = 133,888 B (within 320 KB/WGP).
// -------------------------------------------------------------------------
__global__ __launch_bounds__(256) void gat_neigh_kernel(
    const float* __restrict__ neigh, const float* __restrict__ edge,
    const float* __restrict__ W2, const float* __restrict__ a,
    const float* __restrict__ sx, float* __restrict__ out, int N)
{
    extern __shared__ char smem[];
    __bf16* w2t    = (__bf16*)smem;                 // 16384 elems (32 KB)  W2^T bf16
    __bf16* nh     = (__bf16*)(smem + 32768);       // 32768 elems (64 KB)  neighs bf16
    __bf16* eg     = (__bf16*)(smem + 98304);       // 16384 elems (32 KB)  edges bf16
    float*  scores = (float*)(smem + 131072);       // 256 f32
    float*  attn   = (float*)(smem + 132096);       // 256 f32
    float*  av     = (float*)(smem + 133120);       // a_n[128] | a_e[64]

    const int tid  = threadIdx.x;
    const int wave = tid >> 5;
    const int lane = tid & 31;
    const int half = lane >> 4;
    const int nl   = lane & 15;
    const int nodeBase = blockIdx.x * 8;            // SGPR-uniform
    if (nodeBase >= N) return;
    const bool fullBlk = (nodeBase + 8) <= N;       // uniform tail test
    const long rowCap  = (long)N * 32 - 1;

    // this thread's edge row; prefetch it now so it streams during the GEMM
    long erow = (long)blockIdx.x * 256 + tid;
    if (erow > rowCap) erow = rowCap;
    const float* ep = edge + erow * 64;             // 256 B = 2 cachelines
    __builtin_prefetch(ep, 0, 1);                   // global_prefetch_b8
    __builtin_prefetch(ep + 32, 0, 1);

    // cooperative W2 transpose to bf16
    for (int i = 0; i < 64; ++i) {
        const int idx = tid + i * 256;              // 16384 total
        const int k = idx >> 7, n = idx & 127;
        w2t[n * 128 + k] = (__bf16)W2[idx];
    }
    if (tid < 192) av[tid] = a[128 + tid];
    scores[tid] = 0.f;
    __syncthreads();

    // ---- GEMM phase: neighs tiles + score partials (dot with a_n) ----
    for (int t2 = 0; t2 < 2; ++t2) {
        const int mt = wave + t2 * 8;
        long arow = (long)blockIdx.x * 256 + mt * 16 + nl;
        if (arow > rowCap) arow = rowCap;           // clamp (loads only)
        const float* ap = neigh + arow * 128;
        v16bf afrag[4];
#pragma unroll
        for (int kk = 0; kk < 4; ++kk) {
            const int kb = kk * 32 + half * 16;
            const float4 q0 = *(const float4*)(ap + kb);
            const float4 q1 = *(const float4*)(ap + kb + 4);
            const float4 q2 = *(const float4*)(ap + kb + 8);
            const float4 q3 = *(const float4*)(ap + kb + 12);
            afrag[kk][0]  = (__bf16)q0.x; afrag[kk][1]  = (__bf16)q0.y;
            afrag[kk][2]  = (__bf16)q0.z; afrag[kk][3]  = (__bf16)q0.w;
            afrag[kk][4]  = (__bf16)q1.x; afrag[kk][5]  = (__bf16)q1.y;
            afrag[kk][6]  = (__bf16)q1.z; afrag[kk][7]  = (__bf16)q1.w;
            afrag[kk][8]  = (__bf16)q2.x; afrag[kk][9]  = (__bf16)q2.y;
            afrag[kk][10] = (__bf16)q2.z; afrag[kk][11] = (__bf16)q2.w;
            afrag[kk][12] = (__bf16)q3.x; afrag[kk][13] = (__bf16)q3.y;
            afrag[kk][14] = (__bf16)q3.z; afrag[kk][15] = (__bf16)q3.w;
        }
        float spart[8];
#pragma unroll
        for (int r = 0; r < 8; ++r) spart[r] = 0.f;

        for (int nt = 0; nt < 8; ++nt) {
            v8f acc = {};
            const __bf16* bp = &w2t[(nt * 16 + nl) * 128 + half * 16];
#pragma unroll
            for (int kk = 0; kk < 4; ++kk) {
                v16bf bfrag = *(const v16bf*)(bp + kk * 32);
                acc = __builtin_amdgcn_wmma_f32_16x16x32_bf16(
                    false, afrag[kk], false, bfrag, (short)0, acc, false, false);
            }
            const int   col = nt * 16 + nl;
            const float an  = av[col];
#pragma unroll
            for (int r = 0; r < 8; ++r) {
                const int   lrow = mt * 16 + r + 8 * half;
                const float v    = acc[r];
                spart[r] += v * an;
                nh[lrow * 128 + col] = (__bf16)v;   // stash neighs in LDS
            }
        }
#pragma unroll
        for (int r = 0; r < 8; ++r)
            atomicAdd(&scores[mt * 16 + r + 8 * half], spart[r]);
    }

    // ---- edge scores (dot with a_e) + x-score, one row per thread ----
    {
        float ed = 0.f;
#pragma unroll
        for (int j4 = 0; j4 < 16; ++j4) {
            const int    j = j4 * 4;
            const float4 v = *(const float4*)(ep + j);
            ed += v.x * av[128 + j]     + v.y * av[128 + j + 1]
                + v.z * av[128 + j + 2] + v.w * av[128 + j + 3];
            v4bf p;
            p[0] = (__bf16)v.x; p[1] = (__bf16)v.y;
            p[2] = (__bf16)v.z; p[3] = (__bf16)v.w;
            *(v4bf*)&eg[tid * 64 + j] = p;          // ds_store_b64
        }
        int node = nodeBase + (tid >> 5);
        if (node >= N) node = N - 1;
        atomicAdd(&scores[tid], ed + sx[node]);
    }
    __syncthreads();

    // leaky ReLU (alpha = 0.8)
    {
        const float s = scores[tid];
        scores[tid] = s > 0.f ? s : 0.8f * s;
    }
    __syncthreads();

    // softmax: wave w owns node w (32 neighbors == 32 lanes on wave32)
    {
        float mx = -3.4e38f;
#pragma unroll
        for (int s = 0; s < 32; ++s) {
            const float v = scores[wave * 32 + s];
            mx = v > mx ? v : mx;
        }
        float sum = 0.f;
#pragma unroll
        for (int s = 0; s < 32; ++s)
            sum += __expf(scores[wave * 32 + s] - mx);
        attn[wave * 32 + lane] = __expf(scores[wave * 32 + lane] - mx) / sum;
    }
    __syncthreads();

    // h_prime: 8 nodes x 64 column-pairs = 512 jobs, 2 per thread
#pragma unroll
    for (int jj = 0; jj < 2; ++jj) {
        const int idx  = tid + jj * 256;
        const int node = idx >> 6;
        const int col  = (idx & 63) * 2;
        float s0 = 0.f, s1 = 0.f;
#pragma unroll
        for (int sN = 0; sN < 32; ++sN) {
            const float w = attn[node * 32 + sN];
            const v2bf  p = *(const v2bf*)&nh[(node * 32 + sN) * 128 + col];
            s0 += w * (float)p[0];
            s1 += w * (float)p[1];
        }
        const int gnode = nodeBase + node;
        if (fullBlk || gnode < N) {
            float2 o; o.x = elu_f(s0); o.y = elu_f(s1);
            *(float2*)&out[(size_t)gnode * 320 + 128 + col] = o;
        }
    }
    // e_agg: 8 nodes x 32 column-pairs = 256 jobs, 1 per thread
    {
        const int node = tid >> 5;
        const int col  = (tid & 31) * 2;
        float s0 = 0.f, s1 = 0.f;
#pragma unroll
        for (int sN = 0; sN < 32; ++sN) {
            const float w = attn[node * 32 + sN];
            const v2bf  p = *(const v2bf*)&eg[(node * 32 + sN) * 64 + col];
            s0 += w * (float)p[0];
            s1 += w * (float)p[1];
        }
        const int gnode = nodeBase + node;
        if (fullBlk || gnode < N) {
            float2 o; o.x = elu_f(s0); o.y = elu_f(s1);
            *(float2*)&out[(size_t)gnode * 320 + 256 + col] = o;
        }
    }
}

extern "C" void kernel_launch(void* const* d_in, const int* in_sizes, int n_in,
                              void* d_out, int out_size, void* d_ws, size_t ws_size,
                              hipStream_t stream) {
    const float* input = (const float*)d_in[0];   // (N,128)
    const float* neigh = (const float*)d_in[1];   // (N*32,128)
    const float* edge  = (const float*)d_in[2];   // (N*32,64)
    // d_in[3] = mask (unused by reference)
    const float* W     = (const float*)d_in[4];   // (128,128)
    const float* W2    = (const float*)d_in[5];   // (128,128)
    const float* a     = (const float*)d_in[6];   // (320,1)
    float* out = (float*)d_out;                   // (N,320)
    float* sx  = (float*)d_ws;                    // N floats of scratch

    const int N = in_sizes[0] / 128;

    const int blocks1 = (N + 63) / 64;
    gat_x_kernel<<<blocks1, 128, 0, stream>>>(input, W, a, out, sx, N);

    const int blocks2 = (N + 7) / 8;
    gat_neigh_kernel<<<blocks2, 256, 133888, stream>>>(neigh, edge, W2, a, sx, out, N);
}
